// gnn_gate_9388798509116
// MI455X (gfx1250) — matrix operand order repaired and checked by reading, not compiled, not run on hardware
//
#include <hip/hip_runtime.h>
#include <hip/hip_bf16.h>
#include <math.h>

// ---------------------------------------------------------------------------
// gnn_gate for MI455X (gfx1250, wave32, WMMA 16x16x32 bf16)
//   k0: f32 -> bf16 pre-convert (x, adj, Ww) so GEMM loops carry zero cvt VALU
//   k1: hT[b,o,n] (bf16) = (x @ W^T + Wb) transposed per batch
//   k2: per 16-row tile: h_prime = leakyrelu(adj @ h) via WMMA, fused
//       sigmoid gate + blend; h_prime never leaves LDS.
// ---------------------------------------------------------------------------

typedef __bf16 bf16_t;
typedef bf16_t v16bf __attribute__((ext_vector_type(16)));
typedef float  v8f   __attribute__((ext_vector_type(8)));

union BFrag { unsigned int u[8]; v16bf v; };

__device__ __forceinline__ unsigned int cvt2_bf16(float a, float b) {
    union { bf16_t h[2]; unsigned int u; } r;
    r.h[0] = (bf16_t)a;      // hardware f32->bf16 convert (RNE)
    r.h[1] = (bf16_t)b;
    return r.u;
}

// Two 16B loads of already-bf16 data -> one 16x32 (A) or 32x16 (B) fragment.
__device__ __forceinline__ v16bf load_frag(const bf16_t* __restrict__ p0,
                                           const bf16_t* __restrict__ p1) {
    BFrag f;
    uint4 lo = *reinterpret_cast<const uint4*>(p0);
    uint4 hi = *reinterpret_cast<const uint4*>(p1);
    f.u[0] = lo.x; f.u[1] = lo.y; f.u[2] = lo.z; f.u[3] = lo.w;
    f.u[4] = hi.x; f.u[5] = hi.y; f.u[6] = hi.z; f.u[7] = hi.w;
    return f.v;
}

// ---------------------------------------------------------------------------
// Kernel 0: bulk f32 -> bf16 convert, 8 elements/thread, b128 in / b128 out.
// ---------------------------------------------------------------------------
__global__ __launch_bounds__(256)
void k_cvt_bf16(const float* __restrict__ src, bf16_t* __restrict__ dst) {
    const size_t i = ((size_t)blockIdx.x * 256 + threadIdx.x) * 8;
    float4 a = *reinterpret_cast<const float4*>(src + i);
    float4 b = *reinterpret_cast<const float4*>(src + i + 4);
    uint4 p;
    p.x = cvt2_bf16(a.x, a.y);
    p.y = cvt2_bf16(a.z, a.w);
    p.z = cvt2_bf16(b.x, b.y);
    p.w = cvt2_bf16(b.z, b.w);
    *reinterpret_cast<uint4*>(dst + i) = p;
}

// ---------------------------------------------------------------------------
// Kernel 1: hT (bf16) = (x @ W^T + Wb)^T per batch.
// xB:[8192,256] bf16, WwB:[256,256] bf16 (o,f), Wb:[256] f32, hT:[16,256,512].
// Grid 1024 x 256 threads (8 waves), one 16x16 tile per wave.
// Contiguous k order (A and B): A lanes 0-15 hold K 0-15, lanes 16-31 hold
// K 16-31 (pairs per VGPR); B lane = column, K halves 0-15 / 16-31.
// ---------------------------------------------------------------------------
__global__ __launch_bounds__(256)
void k_h_transpose(const bf16_t* __restrict__ xB,
                   const bf16_t* __restrict__ WwB,
                   const float* __restrict__ Wb,
                   bf16_t* __restrict__ hT) {
    __shared__ float lds[8][256];               // per-wave 16x16 transpose buf

    const int lane    = threadIdx.x & 31;
    const int wv      = threadIdx.x >> 5;
    const int tile    = blockIdx.x * 8 + wv;    // 0..8191
    const int rowTile = tile >> 4;              // 0..511 (16 x-rows each)
    const int oc      = (tile & 15) * 16;       // o-column base

    const int ksel = (lane >> 4) * 16;          // A frag K select
    const int nsel = (lane >> 4) * 16;          // B frag K half select
    const bf16_t* arow = xB + ((size_t)(rowTile * 16 + (lane & 15))) * 256;
    const bf16_t* brow = WwB + ((size_t)(oc + (lane & 15))) * 256;  // B[k,n]=W[n,k]

    v8f c = {};
    #pragma unroll
    for (int kk = 0; kk < 256; kk += 32) {
        v16bf a = load_frag(arow + kk + ksel, arow + kk + ksel + 8);
        v16bf b = load_frag(brow + kk + nsel, brow + kk + nsel + 8);
        c = __builtin_amdgcn_wmma_f32_16x16x32_bf16(
                false, a, false, b, (short)0, c, false, false);
    }

    const float bias = Wb[oc + (lane & 15)];
    #pragma unroll
    for (int r = 0; r < 8; ++r) c[r] += bias;

    // transpose via LDS: C layout VGPR r, lane l -> M = r + 8*(l>>4), N = l&15
    const int o_l = lane & 15;
    #pragma unroll
    for (int r = 0; r < 8; ++r)
        lds[wv][o_l * 16 + (r + 8 * (lane >> 4))] = c[r];
    __syncthreads();

    // coalesced bf16 store: 2 lanes per o-row, 8 elems (16B) each
    const int o_o  = lane >> 1;
    const int half = lane & 1;
    const float* src = &lds[wv][o_o * 16 + half * 8];
    uint4 pk;
    pk.x = cvt2_bf16(src[0], src[1]);
    pk.y = cvt2_bf16(src[2], src[3]);
    pk.z = cvt2_bf16(src[4], src[5]);
    pk.w = cvt2_bf16(src[6], src[7]);
    const int b_idx  = rowTile >> 5;
    const int n_base = (rowTile & 31) * 16 + half * 8;
    bf16_t* dst = hT + ((size_t)(b_idx * 256 + oc + o_o)) * 512 + n_base;
    *reinterpret_cast<uint4*>(dst) = pk;
}

// ---------------------------------------------------------------------------
// Kernel 2: fused adj@h -> leaky_relu -> gate -> blend.
// Grid 512 (b*32 + row-tile), 128 threads (4 waves); wave owns 16 rows x 64
// o-cols (4 WMMA tiles), K loop over 512. h_prime stays in LDS.
// ---------------------------------------------------------------------------
__global__ __launch_bounds__(128)
void k_aggregate_gate(const float* __restrict__ x,        // [16,512,256] f32
                      const bf16_t* __restrict__ adjB,    // [16,512,512] bf16
                      const bf16_t* __restrict__ hT,      // [16,256,512] bf16
                      const float* __restrict__ gate_w,   // [512]
                      const float* __restrict__ gate_b,   // [1]
                      float* __restrict__ out) {          // [16,512,256]
    __shared__ float hp[16][256];
    __shared__ float red[16][8];
    __shared__ float coeff[16];

    const int wg   = blockIdx.x;
    const int b    = wg >> 5;              // batch
    const int nt   = wg & 31;              // 16-row tile within batch
    const int lane = threadIdx.x & 31;
    const int wv   = threadIdx.x >> 5;     // o in [wv*64, wv*64+64)

    const int ksel = (lane >> 4) * 16;
    const int nsel = (lane >> 4) * 16;
    const bf16_t* arow =
        adjB + ((size_t)(b * 512 + nt * 16 + (lane & 15))) * 512;
    const bf16_t* bbase =
        hT + ((size_t)(b * 256 + wv * 64 + (lane & 15))) * 512 + nsel;

    v8f acc[4];
    #pragma unroll
    for (int t = 0; t < 4; ++t) acc[t] = (v8f){};

    #pragma unroll 2
    for (int kk = 0; kk < 512; kk += 32) {
        v16bf a = load_frag(arow + kk + ksel, arow + kk + ksel + 8);
        #pragma unroll
        for (int t = 0; t < 4; ++t) {
            const bf16_t* bp = bbase + (size_t)t * 16 * 512 + kk;
            v16bf bb = load_frag(bp, bp + 8);
            acc[t] = __builtin_amdgcn_wmma_f32_16x16x32_bf16(
                         false, a, false, bb, (short)0, acc[t], false, false);
        }
    }

    // leaky_relu(v, 0.2) == max(v, 0.2*v); park h_prime tile in LDS
    #pragma unroll
    for (int t = 0; t < 4; ++t) {
        const int o = wv * 64 + t * 16 + (lane & 15);
        #pragma unroll
        for (int r = 0; r < 8; ++r) {
            const float v = acc[t][r];
            hp[r + 8 * (lane >> 4)][o] = fmaxf(v, 0.2f * v);
        }
    }
    __syncthreads();

    // gate partials: 8 threads per row, 32 features each over [x | h_prime]
    const int row = threadIdx.x >> 3;
    const int seg = threadIdx.x & 7;
    const float* xrow = x + ((size_t)(b * 512 + nt * 16 + row)) * 256;
    float part = 0.f;
    #pragma unroll 4
    for (int j = 0; j < 32; ++j) {
        const int f = seg * 32 + j;
        part += xrow[f] * gate_w[f] + hp[row][f] * gate_w[256 + f];
    }
    red[row][seg] = part;
    __syncthreads();

    if (threadIdx.x < 16) {
        float s = 0.f;
        #pragma unroll
        for (int j = 0; j < 8; ++j) s += red[threadIdx.x][j];
        s += gate_b[0];
        coeff[threadIdx.x] = 1.f / (1.f + __expf(-s));
    }
    __syncthreads();

    // blend + store: 16x256 elems, 128 threads -> 32 each, coalesced
    #pragma unroll 4
    for (int i = 0; i < 32; ++i) {
        const int idx = i * 128 + threadIdx.x;
        const int r2  = idx >> 8;
        const int c2  = idx & 255;
        const size_t g = ((size_t)(b * 512 + nt * 16 + r2)) * 256 + c2;
        const float cf = coeff[r2];
        out[g] = cf * x[g] + (1.f - cf) * hp[r2][c2];
    }
}

// ---------------------------------------------------------------------------
extern "C" void kernel_launch(void* const* d_in, const int* in_sizes, int n_in,
                              void* d_out, int out_size, void* d_ws, size_t ws_size,
                              hipStream_t stream) {
    const float* x      = (const float*)d_in[0];   // [16,512,256]
    const float* adj    = (const float*)d_in[1];   // [16,512,512]
    const float* Ww     = (const float*)d_in[2];   // [256,256]
    const float* Wb     = (const float*)d_in[3];   // [256]
    // d_in[4] = A : dead code in the reference (DCE'd)
    const float* gate_w = (const float*)d_in[5];   // [1,512]
    const float* gate_b = (const float*)d_in[6];   // [1]
    float* out = (float*)d_out;

    // workspace layout (bytes)
    char* ws = (char*)d_ws;
    bf16_t* xB   = (bf16_t*)(ws);                         //  4 MB  [8192,256]
    bf16_t* adjB = (bf16_t*)(ws + (4u << 20));            //  8 MB  [16,512,512]
    bf16_t* WwB  = (bf16_t*)(ws + (12u << 20) + (600u << 10)); // 128 KB [256,256]
    bf16_t* hT   = (bf16_t*)(ws + (13u << 20));           //  4 MB  [16,256,512]

    // k0: pre-convert to bf16 (adj is 0/1 -> exact)
    k_cvt_bf16<<<(16 * 512 * 256) / (256 * 8), 256, 0, stream>>>(x, xB);
    k_cvt_bf16<<<(16 * 512 * 512) / (256 * 8), 256, 0, stream>>>(adj, adjB);
    k_cvt_bf16<<<(256 * 256) / (256 * 8), 256, 0, stream>>>(Ww, WwB);

    k_h_transpose<<<1024, 256, 0, stream>>>(xB, WwB, Wb, hT);
    k_aggregate_gate<<<512, 128, 0, stream>>>(x, adjB, hT, gate_w, gate_b, out);
}
